// BertTagger_30064771072062
// MI455X (gfx1250) — matrix operand order repaired and checked by reading, not compiled, not run on hardware
//
#include <hip/hip_runtime.h>

// ---------------------------------------------------------------------------
// CRF forward-algorithm (log-space matmul form) for MI455X / gfx1250.
//   alpha_new[b,j] = emit[b,j] + m_b + M_j + log( Ea[b,:] @ Et[j,:]^T )
//   Ea = exp(alpha - rowmax),  Et = exp(trans - rowmax(trans))  (constant!)
// Bulk compute: v_wmma_f32_16x16x32_f16. Et fragments live in VGPRs across
// the whole T=512 loop. Batch tiles are independent per workgroup (no global
// sync). Per-step chain minimized: raw v_exp_f32/v_log_f32, shfl-butterfly
// row max, 2 barriers/step, pipelined ds_load_b128 A-fragments.
// ---------------------------------------------------------------------------

#define CRF_B 512
#define CRF_T 512
#define CRF_K 128
#define CRF_START 126
#define CRF_STOP 127
#define CRF_NEG (-10000.0f)

#define LOG2E 1.4426950408889634f
#define LN2   0.6931471805599453f

typedef __attribute__((ext_vector_type(16))) _Float16 v16h;
typedef __attribute__((ext_vector_type(8)))  _Float16 v8h;
typedef __attribute__((ext_vector_type(8)))  float    v8f;

// ---------------------------------------------------------------------------
// Prep: row maxes of trans, stop-row copy, Et stored k-major as f16:
//   Bk[j*128 + k] = exp(trans[j,k] - M_j)   (j-major: WMMA B column per lane)
// ---------------------------------------------------------------------------
__global__ __launch_bounds__(256) void crf_prep_kernel(
    const float* __restrict__ trans, float* __restrict__ Mrow,
    float* __restrict__ tStop, _Float16* __restrict__ Bk) {
  const int tid = threadIdx.x;
  if (tid < CRF_K) {
    float m = -3.0e38f;
    for (int k = 0; k < CRF_K; ++k) m = fmaxf(m, trans[tid * CRF_K + k]);
    Mrow[tid]  = m;
    tStop[tid] = trans[CRF_STOP * CRF_K + tid];
  }
  __syncthreads();
  for (int idx = tid; idx < CRF_K * CRF_K; idx += 256) {
    const int j = idx >> 7, k = idx & (CRF_K - 1);
    Bk[idx] = (_Float16)__builtin_amdgcn_exp2f(
        (trans[j * CRF_K + k] - Mrow[j]) * LOG2E);
  }
}

// ---------------------------------------------------------------------------
// Forward recurrence: one block = 16 batch rows, 8 waves = 8 column tiles.
// ---------------------------------------------------------------------------
__global__ __launch_bounds__(256) void crf_forward_kernel(
    const float* __restrict__ feats, const int* __restrict__ lengths,
    const float* __restrict__ Mrow, const float* __restrict__ tStop,
    const _Float16* __restrict__ Bk, float* __restrict__ fwd) {
  __shared__ float    alpha[16][132];   // stride 132 -> bank stagger
  __shared__ _Float16 Ea[16][136];      // stride 136 halves = 272B (16B mult)
  __shared__ float    mrow[16];
  __shared__ float    tS[CRF_K];
  __shared__ int      llen[16];

  const int tid   = threadIdx.x;
  const int lane  = tid & 31;
  const int wave  = tid >> 5;          // 0..7: output column tile
  const int nrow  = lane & 15;
  const int half  = lane >> 4;
  const int Bbase = blockIdx.x * 16;
  const int j     = wave * 16 + nrow;  // D-matrix: N = lane % 16

  const float Mj = Mrow[j];

  // Constant Et fragments: B 32x16 f16 frag (k order chosen here, same as A): lane holds column n = lane%16,
  // 16 contiguous k values (k = 16*half + s) -> one aligned 32B load per
  // K-block. Held in VGPRs across the entire time loop.
  v16h Bf[4];
#pragma unroll
  for (int kb = 0; kb < 4; ++kb)
    Bf[kb] = *(const v16h*)(Bk + j * CRF_K + kb * 32 + half * 16);

  if (tid < CRF_K) tS[tid] = tStop[tid];
  if (tid < 16)    llen[tid] = lengths[Bbase + tid];

  const int r = tid >> 4, c = tid & 15;   // this thread's alpha row / chunk
  {
#pragma unroll
    for (int i = 0; i < 8; ++i) {
      const int col = c * 8 + i;
      alpha[r][col] = (col == CRF_START) ? 0.0f : CRF_NEG;
    }
  }
  __syncthreads();

  const float* featp = feats + (size_t)(Bbase + 8 * half) * CRF_T * CRF_K + j;

  for (int t = 0; t < CRF_T; ++t) {
    // --- issue emissions early; latency hides behind max/exp/wmma ---
    float emitv[8];
#pragma unroll
    for (int v = 0; v < 8; ++v)
      emitv[v] = featp[(size_t)v * CRF_T * CRF_K + (size_t)t * CRF_K];

    // --- per-batch-row max via in-half-wave butterfly (no barrier) ---
    float av[8];
    float mb = -3.0e38f;
#pragma unroll
    for (int i = 0; i < 8; ++i) {
      av[i] = alpha[r][c * 8 + i];
      mb = fmaxf(mb, av[i]);
    }
    mb = fmaxf(mb, __shfl_xor(mb, 1, 32));
    mb = fmaxf(mb, __shfl_xor(mb, 2, 32));
    mb = fmaxf(mb, __shfl_xor(mb, 4, 32));
    mb = fmaxf(mb, __shfl_xor(mb, 8, 32));
    if ((lane & 15) == 0) mrow[r] = mb;   // visible after the next barrier

    // --- Ea = exp(alpha - m) as f16 in LDS (raw v_exp_f32) ---
#pragma unroll
    for (int i = 0; i < 8; ++i)
      Ea[r][c * 8 + i] =
          (_Float16)__builtin_amdgcn_exp2f((av[i] - mb) * LOG2E);
    __syncthreads();

    // --- S = Ea @ Et^T via WMMA: preload all A-frags, then MAC chain ---
    v16h afr[4];
#pragma unroll
    for (int kb = 0; kb < 4; ++kb) {
      // A 16x32 f16: lane holds M = lane%16; K chunk split by lane-half.
      const _Float16* p = &Ea[nrow][kb * 32 + half * 16];
      v8h c0 = *(const v8h*)p;         // K = 32*kb + 16*half .. +7
      v8h c1 = *(const v8h*)(p + 8);   // K = 32*kb + 16*half + 8 .. +15
#pragma unroll
      for (int i = 0; i < 8; ++i) { afr[kb][i] = c0[i]; afr[kb][i + 8] = c1[i]; }
    }
    v8f acc = {};
#pragma unroll
    for (int kb = 0; kb < 4; ++kb)
      acc = __builtin_amdgcn_wmma_f32_16x16x32_f16(
          false, afr[kb], false, Bf[kb], (short)0, acc, false, false);

    // --- alpha_new = emit + log(S) + M_j + m_b (raw v_log_f32) ---
#pragma unroll
    for (int v = 0; v < 8; ++v) {
      const int row = v + 8 * half;                 // D: M = v + 8*(lane>=16)
      const float S = fmaxf(acc[v], 1e-30f);        // stays in normal range
      alpha[row][j] =
          emitv[v] + __builtin_amdgcn_logf(S) * LN2 + Mj + mrow[row];
    }
    __syncthreads();

    // --- fold in the lengths-indexed terminal LSE (fires once per row) ---
    if (tid < 16 && llen[tid] == t + 1) {
      float mm = -3.0e38f;
      for (int jj = 0; jj < CRF_K; ++jj)
        mm = fmaxf(mm, alpha[tid][jj] + tS[jj]);
      float s = 0.0f;
      for (int jj = 0; jj < CRF_K; ++jj)
        s += __builtin_amdgcn_exp2f((alpha[tid][jj] + tS[jj] - mm) * LOG2E);
      fwd[Bbase + tid] = mm + __builtin_amdgcn_logf(s) * LN2;
    }
    // alpha is next written only after the next barrier -> no extra sync
  }
}

// ---------------------------------------------------------------------------
// Gold path score: one thread per batch element (tiny: 262K terms total).
// ---------------------------------------------------------------------------
__global__ __launch_bounds__(256) void crf_gold_kernel(
    const float* __restrict__ feats, const float* __restrict__ trans,
    const int* __restrict__ tags, const int* __restrict__ lengths,
    float* __restrict__ gold) {
  const int b = blockIdx.x * blockDim.x + threadIdx.x;
  if (b >= CRF_B) return;
  const int len = lengths[b];
  int prev = CRF_START;
  float score = 0.0f;
  for (int t = 0; t < len; ++t) {
    const int tg = tags[b * CRF_T + t];
    score += trans[tg * CRF_K + prev];                       // trans[ps, pa]
    score += feats[((size_t)b * CRF_T + t) * CRF_K + tg];    // emission
    prev = tg;
  }
  score += trans[CRF_STOP * CRF_K + prev];                   // -> STOP
  gold[b] = score;
}

// ---------------------------------------------------------------------------
// mean |forward - gold| over B = 512.
// ---------------------------------------------------------------------------
__global__ __launch_bounds__(256) void crf_reduce_kernel(
    const float* __restrict__ fwd, const float* __restrict__ gold,
    float* __restrict__ out) {
  __shared__ float sm[256];
  const int tid = threadIdx.x;
  sm[tid] = fabsf(fwd[tid] - gold[tid]) +
            fabsf(fwd[tid + 256] - gold[tid + 256]);
  __syncthreads();
  for (int off = 128; off > 0; off >>= 1) {
    if (tid < off) sm[tid] += sm[tid + off];
    __syncthreads();
  }
  if (tid == 0) out[0] = sm[0] * (1.0f / (float)CRF_B);
}

// ---------------------------------------------------------------------------
extern "C" void kernel_launch(void* const* d_in, const int* in_sizes, int n_in,
                              void* d_out, int out_size, void* d_ws,
                              size_t ws_size, hipStream_t stream) {
  (void)in_sizes; (void)n_in; (void)out_size; (void)ws_size;
  const float* feats   = (const float*)d_in[0];  // [B,T,K] f32
  const float* trans   = (const float*)d_in[1];  // [K,K]   f32
  const int*   tags    = (const int*)d_in[2];    // [B,T]   i32
  const int*   lengths = (const int*)d_in[3];    // [B]     i32
  float*       out     = (float*)d_out;          // scalar f32

  char* w = (char*)d_ws;
  float*    Mrow  = (float*)(w);                         // 128 f32
  float*    tStop = (float*)(w + 512);                   // 128 f32
  _Float16* Bk    = (_Float16*)(w + 1024);               // 128*128 f16 (32KB)
  float*    fwd   = (float*)(w + 1024 + 32768);          // 512 f32
  float*    gold  = (float*)(w + 1024 + 32768 + 2048);   // 512 f32

  crf_prep_kernel<<<1, 256, 0, stream>>>(trans, Mrow, tStop, Bk);
  crf_forward_kernel<<<CRF_B / 16, 256, 0, stream>>>(feats, lengths, Mrow,
                                                     tStop, Bk, fwd);
  crf_gold_kernel<<<CRF_B / 256, 256, 0, stream>>>(feats, trans, tags,
                                                   lengths, gold);
  crf_reduce_kernel<<<1, 256, 0, stream>>>(fwd, gold, out);
}
